// Decoder_17162689314960
// MI455X (gfx1250) — matrix operand order repaired and checked by reading, not compiled, not run on hardware
//
#include <hip/hip_runtime.h>

// ---------------------------------------------------------------------------
// Tacotron-2 style decoder for MI455X (gfx1250), wave32 + WMMA bf16.
// Persistent kernel, 64 WGs x 256 threads, global sense-reversing barrier.
// Recurrent GEMMs on v_wmma_f32_16x16x32_bf16 (f32 accumulate); A-panels are
// staged into LDS with global_load_async_to_lds_b128 (ASYNCcnt path).
// ---------------------------------------------------------------------------

typedef __attribute__((ext_vector_type(16))) __bf16 v16bf;
typedef __attribute__((ext_vector_type(8)))  __bf16 v8bf;
typedef __attribute__((ext_vector_type(8)))  float  v8f;

union AF { v16bf v; v8bf h[2]; };

#define WMMA_BF16(a, b, c) \
  __builtin_amdgcn_wmma_f32_16x16x32_bf16(false, (a), false, (b), (short)0, (c), false, false)

// Problem constants
constexpr int B = 64, T = 300, TM = 400, D = 512, H = 512, A = 128, NMEL = 80;
constexpr int NWG = 64;

// ---- workspace layout (bytes) ----------------------------------------------
constexpr size_t OFF_BAR   = 0;                      // 2 x u32 barrier state
constexpr size_t OFF_AH    = 256;                    // (64,512) f32
constexpr size_t OFF_AC    = OFF_AH   + 131072;
constexpr size_t OFF_DH    = OFF_AC   + 131072;
constexpr size_t OFF_DC    = OFF_DH   + 131072;
constexpr size_t OFF_CTX   = OFF_DC   + 131072;      // (64,512) f32
constexpr size_t OFF_Q     = OFF_CTX  + 131072;      // (64,128) f32
constexpr size_t OFF_GATES = OFF_Q    + 32768;       // (64,2048) f32
constexpr size_t OFF_Z1    = OFF_GATES + 524288;     // (64,1152) bf16 [x|ctx|ah]
constexpr size_t OFF_Z2    = OFF_Z1   + 147456;      // (64,1536) bf16 [ah|ctx|dh]
constexpr size_t OFF_Z3    = OFF_Z2   + 196608;      // (64,1024) bf16 [dh|ctx]
constexpr size_t OFF_STATE_END = OFF_Z3 + 131072;    // everything above re-zeroed per call
constexpr size_t OFF_WARNN = OFF_STATE_END;          // packed bf16, K=1152,N=2048
constexpr size_t OFF_WDRNN = OFF_WARNN + 4718592;    // packed bf16, K=1536,N=2048
constexpr size_t OFF_WQ    = OFF_WDRNN + 6291456;    // packed bf16, K=512,N=128
constexpr size_t OFF_WM    = OFF_WQ    + 131072;     // packed bf16, K=512,N=128
constexpr size_t OFF_WMEL  = OFF_WM    + 131072;     // packed bf16, K=1024,N=96
constexpr size_t OFF_MEMBF = OFF_WMEL  + 196608;     // memory as bf16 (64*400,512)
constexpr size_t OFF_MPROJ = OFF_MEMBF + 26214400;   // m_proj f32 (64*400,128)

// ---- output layout (f32) ---------------------------------------------------
constexpr size_t OUT_MEL  = 0;                       // (64,80,300)
constexpr size_t OUT_STOP = 1536000;                 // (64,300)
constexpr size_t OUT_ATTN = 1555200;                 // (64,300,400)

// ---------------------------------------------------------------------------
__device__ inline __bf16 f2bf(float f) {             // round-to-nearest-even
  unsigned u = __float_as_uint(f);
  unsigned short s = (unsigned short)((u + 0x7FFFu + ((u >> 16) & 1u)) >> 16);
  return __builtin_bit_cast(__bf16, s);
}
__device__ inline float bf2f(__bf16 v) {
  unsigned short s = __builtin_bit_cast(unsigned short, v);
  return __uint_as_float(((unsigned)s) << 16);
}
__device__ inline float sigm(float x) { return 1.0f / (1.0f + expf(-x)); }

// Async 16B global -> LDS copy (per-lane), CDNA5 GLOBAL_LOAD_ASYNC_TO_LDS_B128.
// VDST = per-lane LDS byte address, VADDR = 64-bit global address (GV mode).
__device__ inline void async_copy_b128(const void* gsrc, void* ldst) {
  typedef __attribute__((address_space(3))) unsigned char lds_uc;
  unsigned laddr = (unsigned)(unsigned long long)(lds_uc*)ldst;  // 32-bit LDS offset
  unsigned long long gaddr = (unsigned long long)gsrc;
  asm volatile("global_load_async_to_lds_b128 %0, %1, off"
               :: "v"(laddr), "v"(gaddr) : "memory");
}
__device__ inline void async_wait0() {
  asm volatile("s_wait_asynccnt 0" ::: "memory");
}

// packed-weight index decode: element p -> (k, n) given KT (K/32)
__device__ inline void pk_idx(long p, int KT, int& k, int& n) {
  int e  = (int)(p & 15);
  int ln = (int)((p >> 4) & 31);
  long f = p >> 9;
  int kt = (int)(f % KT), nt = (int)(f / KT);
  k = kt * 32 + (e >> 3) * 16 + (ln >> 4) * 8 + (e & 7);
  n = nt * 16 + (ln & 15);
}

// A fragment (16-bit layout): row = lane%16, two 16B chunks at K = grp*8, 16+grp*8
__device__ inline v16bf a_frag(const __bf16* base, int ld, int m0, int k0, int lane) {
  int row = m0 + (lane & 15), grp = lane >> 4;
  const __bf16* ap = base + (size_t)row * ld + k0 + grp * 8;
  AF a;
  a.h[0] = *(const v8bf*)ap;
  a.h[1] = *(const v8bf*)(ap + 16);
  return a.v;
}

// grid-wide barrier (sense via generation counter, agent scope)
__device__ inline void gbar(unsigned* bar) {
  __syncthreads();
  if (threadIdx.x == 0) {
    unsigned gen = __hip_atomic_load(bar + 1, __ATOMIC_RELAXED, __HIP_MEMORY_SCOPE_AGENT);
    unsigned v = __hip_atomic_fetch_add(bar, 1u, __ATOMIC_ACQ_REL, __HIP_MEMORY_SCOPE_AGENT);
    if (v == NWG - 1) {
      __hip_atomic_store(bar, 0u, __ATOMIC_RELAXED, __HIP_MEMORY_SCOPE_AGENT);
      __hip_atomic_fetch_add(bar + 1, 1u, __ATOMIC_RELEASE, __HIP_MEMORY_SCOPE_AGENT);
    } else {
      while (__hip_atomic_load(bar + 1, __ATOMIC_ACQUIRE, __HIP_MEMORY_SCOPE_AGENT) == gen)
        __builtin_amdgcn_s_sleep(1);
    }
  }
  __syncthreads();
}

// Gate GEMM: (64 x K) bf16 @ packed (K x 2048) -> (64 x 2048) f32.
// WG async-stages its 16-row A panel in LDS; each of the 8 waves owns a 16x16 tile.
__device__ inline void gates_gemm(const __bf16* z, const v16bf* pw, float* gout,
                                  int K, unsigned char* smem) {
  const int tid = threadIdx.x, lane = tid & 31, wv = tid >> 5, wg = blockIdx.x;
  const int mt = wg >> 4;
  const int nt = (wg & 15) * 8 + wv;
  const int bytes = 16 * K * 2;
  const char* src = (const char*)(z + (size_t)mt * 16 * K);   // rows contiguous
  for (int off = tid * 16; off < bytes; off += 256 * 16)
    async_copy_b128(src + off, smem + off);
  async_wait0();
  __syncthreads();
  const __bf16* sa = (const __bf16*)smem;
  const int KT = K >> 5, row = lane & 15, grp = lane >> 4;
  v8f acc = {};
  for (int kt = 0; kt < KT; ++kt) {
    AF a;
    const __bf16* ap = sa + (size_t)row * K + kt * 32 + grp * 8;
    a.h[0] = *(const v8bf*)ap;
    a.h[1] = *(const v8bf*)(ap + 16);
    if (kt + 4 < KT)
      __builtin_prefetch(&pw[(size_t)(nt * KT + kt + 4) * 32 + lane], 0, 0);
    v16bf bf = pw[(size_t)(nt * KT + kt) * 32 + lane];
    acc = WMMA_BF16(a.v, bf, acc);
  }
  int col = nt * 16 + (lane & 15), r0 = mt * 16 + grp * 8;
#pragma unroll
  for (int r = 0; r < 8; ++r) gout[(size_t)(r0 + r) * 2048 + col] = acc[r];
}

// LSTM pointwise: gates(64,2048 f32, order i|f|g|o) -> h,c + bf16 copies
__device__ inline void lstm_pw(const float* gates, const float* bih, const float* bhh,
                               float* hbuf, float* cbuf,
                               __bf16* d1, int ld1, __bf16* d2, int ld2) {
  int gtid = blockIdx.x * 256 + threadIdx.x;
  for (int idx = gtid; idx < B * H; idx += NWG * 256) {
    int b = idx >> 9, j = idx & 511;
    const float* gb = gates + (size_t)b * 2048;
    float gi = gb[j]        + bih[j]        + bhh[j];
    float gf = gb[512 + j]  + bih[512 + j]  + bhh[512 + j];
    float gg = gb[1024 + j] + bih[1024 + j] + bhh[1024 + j];
    float go = gb[1536 + j] + bih[1536 + j] + bhh[1536 + j];
    float c = sigm(gf) * cbuf[idx] + sigm(gi) * tanhf(gg);
    float h = sigm(go) * tanhf(c);
    cbuf[idx] = c; hbuf[idx] = h;
    __bf16 hb = f2bf(h);
    d1[(size_t)b * ld1 + j] = hb;
    d2[(size_t)b * ld2 + j] = hb;
  }
}

// ---------------------------------------------------------------------------
__global__ void k_init(unsigned char* ws) {
  unsigned* p = (unsigned*)ws;
  size_t nwords = OFF_STATE_END / 4;
  for (size_t i = blockIdx.x * blockDim.x + threadIdx.x; i < nwords;
       i += (size_t)gridDim.x * blockDim.x)
    p[i] = 0;
}

// fragment-swizzle weights to bf16 + convert memory to bf16
__global__ void k_pack(const float* memory, const float* awih, const float* awhh,
                       const float* dwih, const float* dwhh, const float* wq,
                       const float* wm, const float* melw, unsigned char* ws) {
  __bf16* membf = (__bf16*)(ws + OFF_MEMBF);
  __bf16* pa   = (__bf16*)(ws + OFF_WARNN);
  __bf16* pd   = (__bf16*)(ws + OFF_WDRNN);
  __bf16* pq   = (__bf16*)(ws + OFF_WQ);
  __bf16* pm   = (__bf16*)(ws + OFF_WM);
  __bf16* pmel = (__bf16*)(ws + OFF_WMEL);
  const long NMEM = 13107200, NA = 2359296, ND = 3145728, NQ = 65536, NW = 65536, NL = 98304;
  const long total = NMEM + NA + ND + NQ + NW + NL;
  for (long i = (long)blockIdx.x * blockDim.x + threadIdx.x; i < total;
       i += (long)gridDim.x * blockDim.x) {
    long p = i; int k, n;
    if (p < NMEM) { membf[p] = f2bf(memory[p]); continue; }
    p -= NMEM;
    if (p < NA) { pk_idx(p, 36, k, n);
      pa[p] = f2bf(k < 640 ? awih[(size_t)n * 640 + k] : awhh[(size_t)n * 512 + (k - 640)]);
      continue; }
    p -= NA;
    if (p < ND) { pk_idx(p, 48, k, n);
      pd[p] = f2bf(k < 1024 ? dwih[(size_t)n * 1024 + k] : dwhh[(size_t)n * 512 + (k - 1024)]);
      continue; }
    p -= ND;
    if (p < NQ) { pk_idx(p, 16, k, n); pq[p] = f2bf(wq[(size_t)n * 512 + k]); continue; }
    p -= NQ;
    if (p < NW) { pk_idx(p, 16, k, n); pm[p] = f2bf(wm[(size_t)n * 512 + k]); continue; }
    p -= NW;
    pk_idx(p, 32, k, n);
    pmel[p] = f2bf(n < NMEL ? melw[(size_t)n * 1024 + k] : 0.0f);
  }
}

// m_proj = memory(bf16, 25600x512) @ wm^T(512x128), WMMA, one 16x16 tile / wave
__global__ void k_mproj(unsigned char* ws) {
  const __bf16* membf = (const __bf16*)(ws + OFF_MEMBF);
  const v16bf*  pw    = (const v16bf*)(ws + OFF_WM);
  float* mproj = (float*)(ws + OFF_MPROJ);
  int lane = threadIdx.x & 31;
  int gw = blockIdx.x * 8 + (threadIdx.x >> 5);
  if (gw >= 1600 * 8) return;
  int mt = gw >> 3, nt = gw & 7, grp = lane >> 4;
  v8f acc = {};
  for (int kt = 0; kt < 16; ++kt) {
    v16bf a = a_frag(membf, 512, mt * 16, kt * 32, lane);
    v16bf b = pw[(size_t)(nt * 16 + kt) * 32 + lane];
    acc = WMMA_BF16(a, b, acc);
  }
  int col = nt * 16 + (lane & 15), r0 = mt * 16 + grp * 8;
#pragma unroll
  for (int r = 0; r < 8; ++r) mproj[(size_t)(r0 + r) * 128 + col] = acc[r];
}

// ---------------------------------------------------------------------------
__global__ void __launch_bounds__(256, 1)
k_decoder(unsigned char* ws, const float* memory, const float* mels,
          const float* pre_w1, const float* pre_b1, const float* pre_w2, const float* pre_b2,
          const float* arnn_bih, const float* arnn_bhh, const float* attn_v,
          const float* drnn_bih, const float* drnn_bhh,
          const float* mel_b, const float* stop_w, const float* stop_bp,
          const unsigned char* mask, float* out) {
  __shared__ __align__(16) unsigned char smem[49152];
  unsigned* bar = (unsigned*)(ws + OFF_BAR);
  float* ah   = (float*)(ws + OFF_AH);
  float* ac   = (float*)(ws + OFF_AC);
  float* dh   = (float*)(ws + OFF_DH);
  float* dc   = (float*)(ws + OFF_DC);
  float* ctxf = (float*)(ws + OFF_CTX);
  float* qbuf = (float*)(ws + OFF_Q);
  float* gates = (float*)(ws + OFF_GATES);
  __bf16* z1 = (__bf16*)(ws + OFF_Z1);   // [x(128)|ctx(512)|ah(512)]
  __bf16* z2 = (__bf16*)(ws + OFF_Z2);   // [ah(512)|ctx(512)|dh(512)]
  __bf16* z3 = (__bf16*)(ws + OFF_Z3);   // [dh(512)|ctx(512)]
  const v16bf* pwA   = (const v16bf*)(ws + OFF_WARNN);
  const v16bf* pwD   = (const v16bf*)(ws + OFF_WDRNN);
  const v16bf* pwQ   = (const v16bf*)(ws + OFF_WQ);
  const v16bf* pwMEL = (const v16bf*)(ws + OFF_WMEL);
  const float* mproj = (const float*)(ws + OFF_MPROJ);

  const int tid = threadIdx.x, lane = tid & 31, wv = tid >> 5, wg = blockIdx.x;
  const int gwave = wg * 8 + wv;
  const float stop_b0 = stop_bp[0];

  for (int t = 0; t < T; ++t) {
    // -- phase 1: prenet (WG0 only); x -> z1[:, 0:128] as bf16 ---------------
    if (wg == 0) {
      __bf16* h1 = (__bf16*)smem;                 // (64,256) bf16 = 32KB
      int u = tid;                                 // layer-1 unit
      float b1v = pre_b1[u];
      for (int b = 0; b < B; ++b) {
        float s = b1v;
        if (t > 0) {
          const float* pm = mels + (size_t)b * NMEL * T + (t - 1);
          for (int m = 0; m < NMEL; ++m) s += pm[(size_t)m * T] * pre_w1[u * NMEL + m];
        }
        h1[b * 256 + u] = f2bf(fmaxf(s, 0.0f));
      }
      __syncthreads();
      int u2 = tid & 127, bg = tid >> 7;           // layer-2: 2 half-batches
      for (int b = bg * 32; b < bg * 32 + 32; ++b) {
        float s = pre_b2[u2];
        for (int m = 0; m < 256; ++m) s += bf2f(h1[b * 256 + m]) * pre_w2[u2 * 256 + m];
        z1[(size_t)b * 1152 + u2] = f2bf(fmaxf(s, 0.0f));
      }
    }
    gbar(bar);

    // -- phase 2: attention-LSTM gates GEMM (K=1152) -------------------------
    gates_gemm(z1, pwA, gates, 1152, smem);
    gbar(bar);

    // -- phase 3: attention-LSTM pointwise -> ah,ac; bf16 into z1.ah / z2.ah --
    lstm_pw(gates, arnn_bih, arnn_bhh, ah, ac, z1 + 640, 1152, z2 + 0, 1536);
    gbar(bar);

    // -- phase 4: q = ah @ wq^T  (M=64,K=512,N=128), 32 waves ----------------
    if (gwave < 32) {
      int mt = gwave >> 3, nt = gwave & 7, grp = lane >> 4;
      v8f acc = {};
      for (int kt = 0; kt < 16; ++kt) {
        v16bf a = a_frag(z2, 1536, mt * 16, kt * 32, lane);   // z2[:,0:512] = ah
        v16bf b = pwQ[(size_t)(nt * 16 + kt) * 32 + lane];
        acc = WMMA_BF16(a, b, acc);
      }
      int col = nt * 16 + (lane & 15), r0 = mt * 16 + grp * 8;
#pragma unroll
      for (int r = 0; r < 8; ++r) qbuf[(size_t)(r0 + r) * 128 + col] = acc[r];
    }
    gbar(bar);

    // -- phase 5: Bahdanau scores + softmax + context (wave per batch row) ---
    if (gwave < B) {
      int b = gwave;
      float* wsm = (float*)smem + wv * 512;        // per-wave weights buffer
      float sc[13];
      float lmax = -3.4e38f;
      for (int j = 0; j < 13; ++j) {
        int tm = lane + j * 32;
        float s = -3.4e38f;
        if (tm < TM) {
          s = 0.0f;
          const float* mp = mproj + ((size_t)(b * TM + tm)) * A;
          const float* qr = qbuf + (size_t)b * A;
          for (int a = 0; a < A; ++a) s += tanhf(qr[a] + mp[a]) * attn_v[a];
          if (!mask[b * TM + tm]) s -= 1e9f;
        }
        sc[j] = s;
        lmax = fmaxf(lmax, s);
      }
      for (int m = 16; m; m >>= 1) lmax = fmaxf(lmax, __shfl_xor(lmax, m, 32));
      float lsum = 0.0f;
      for (int j = 0; j < 13; ++j) {
        int tm = lane + j * 32;
        float e = (tm < TM) ? expf(sc[j] - lmax) : 0.0f;
        sc[j] = e; lsum += e;
      }
      for (int m = 16; m; m >>= 1) lsum += __shfl_xor(lsum, m, 32);
      float inv = 1.0f / lsum;
      for (int j = 0; j < 13; ++j) {
        int tm = lane + j * 32;
        if (tm < TM) {
          float w = sc[j] * inv;
          wsm[tm] = w;
          out[OUT_ATTN + ((size_t)(b * T + t)) * TM + tm] = w;
        }
      }
      // ctx[b,d] = sum_tm w[tm] * memory[b,tm,d]
      for (int jj = 0; jj < 16; ++jj) {
        int d = lane + jj * 32;
        float acc = 0.0f;
        const float* mb = memory + ((size_t)b * TM) * D + d;
        for (int tm = 0; tm < TM; ++tm) acc += wsm[tm] * mb[(size_t)tm * D];
        ctxf[b * D + d] = acc;
        __bf16 cb = f2bf(acc);
        z1[(size_t)b * 1152 + 128 + d] = cb;
        z2[(size_t)b * 1536 + 512 + d] = cb;
        z3[(size_t)b * 1024 + 512 + d] = cb;
      }
    }
    gbar(bar);

    // -- phase 6: decoder-LSTM gates GEMM (K=1536) ---------------------------
    gates_gemm(z2, pwD, gates, 1536, smem);
    gbar(bar);

    // -- phase 7: decoder-LSTM pointwise -> dh,dc; bf16 into z2.dh / z3.dh ----
    lstm_pw(gates, drnn_bih, drnn_bhh, dh, dc, z2 + 1024, 1536, z3 + 0, 1024);
    gbar(bar);

    // -- phase 8: mel GEMM (K=1024, N=96 padded) + stop logits ---------------
    if (gwave < 24) {
      int mt = gwave / 6, nt = gwave % 6, grp = lane >> 4;
      v8f acc = {};
      for (int kt = 0; kt < 32; ++kt) {
        v16bf a = a_frag(z3, 1024, mt * 16, kt * 32, lane);
        v16bf b = pwMEL[(size_t)(nt * 32 + kt) * 32 + lane];
        acc = WMMA_BF16(a, b, acc);
      }
      int col = nt * 16 + (lane & 15), r0 = mt * 16 + grp * 8;
      if (col < NMEL) {
        float bias = mel_b[col];
#pragma unroll
        for (int r = 0; r < 8; ++r)
          out[OUT_MEL + ((size_t)(r0 + r) * NMEL + col) * T + t] = acc[r] + bias;
      }
    } else if (gwave < 32) {
      int i = gwave - 24;
      for (int bb = 0; bb < 8; ++bb) {
        int b = i * 8 + bb;
        float s = 0.0f;
        for (int k = lane; k < 1024; k += 32) {
          float pv = (k < 512) ? dh[b * D + k] : ctxf[b * D + (k - 512)];
          s += pv * stop_w[k];
        }
        for (int m = 16; m; m >>= 1) s += __shfl_xor(s, m, 32);
        if (lane == 0) out[OUT_STOP + (size_t)b * T + t] = s + stop_b0;
      }
    }
    gbar(bar);   // step boundary (prenet(t+1) only touches z1.x)
  }
}

// ---------------------------------------------------------------------------
extern "C" void kernel_launch(void* const* d_in, const int* in_sizes, int n_in,
                              void* d_out, int out_size, void* d_ws, size_t ws_size,
                              hipStream_t stream) {
  (void)in_sizes; (void)n_in; (void)out_size; (void)ws_size;
  unsigned char* ws = (unsigned char*)d_ws;
  const float* memory    = (const float*)d_in[0];
  const float* mels      = (const float*)d_in[1];
  const float* pre_w1    = (const float*)d_in[2];
  const float* pre_b1    = (const float*)d_in[3];
  const float* pre_w2    = (const float*)d_in[4];
  const float* pre_b2    = (const float*)d_in[5];
  const float* arnn_wih  = (const float*)d_in[6];
  const float* arnn_whh  = (const float*)d_in[7];
  const float* arnn_bih  = (const float*)d_in[8];
  const float* arnn_bhh  = (const float*)d_in[9];
  const float* attn_wq   = (const float*)d_in[10];
  const float* attn_wm   = (const float*)d_in[11];
  const float* attn_v    = (const float*)d_in[12];
  const float* drnn_wih  = (const float*)d_in[13];
  const float* drnn_whh  = (const float*)d_in[14];
  const float* drnn_bih  = (const float*)d_in[15];
  const float* drnn_bhh  = (const float*)d_in[16];
  const float* mel_w     = (const float*)d_in[17];
  const float* mel_b     = (const float*)d_in[18];
  const float* stop_w    = (const float*)d_in[19];
  const float* stop_b    = (const float*)d_in[20];
  const unsigned char* memory_mask = (const unsigned char*)d_in[21];

  k_init<<<256, 256, 0, stream>>>(ws);
  k_pack<<<2048, 256, 0, stream>>>(memory, arnn_wih, arnn_whh, drnn_wih, drnn_whh,
                                   attn_wq, attn_wm, mel_w, ws);
  k_mproj<<<1600, 256, 0, stream>>>(ws);
  k_decoder<<<NWG, 256, 0, stream>>>(ws, memory, mels,
                                     pre_w1, pre_b1, pre_w2, pre_b2,
                                     arnn_bih, arnn_bhh, attn_v,
                                     drnn_bih, drnn_bhh,
                                     mel_b, stop_w, stop_b,
                                     memory_mask, (float*)d_out);
}
